// ScaledDotProductAttentionLayer_63608465654089
// MI455X (gfx1250) — matrix operand order repaired and checked by reading, not compiled, not run on hardware
//
#include <hip/hip_runtime.h>

#define BQ 4
#define SQ 4096
#define DM 512
#define DKV 64

typedef __attribute__((ext_vector_type(16))) __bf16 v16bf;
typedef __attribute__((ext_vector_type(8)))  float  v8f;
typedef __attribute__((ext_vector_type(8)))  unsigned int v8u;
typedef __attribute__((__vector_size__(4 * sizeof(int)))) int v4i_;

__device__ __forceinline__ unsigned short f2bf(float f) {
  unsigned int u = __float_as_uint(f);
  u += 0x7FFFu + ((u >> 16) & 1u);   // round-to-nearest-even
  return (unsigned short)(u >> 16);
}
__device__ __forceinline__ unsigned int pack2(float a, float b) {
  return (unsigned int)f2bf(a) | ((unsigned int)f2bf(b) << 16);
}

// ---- CDNA5 async copy global->LDS (ASYNCcnt path), 16 bytes per lane ----
__device__ __forceinline__ void async16(const unsigned int* g, unsigned int* l) {
#if __has_builtin(__builtin_amdgcn_global_load_async_to_lds_b128)
  __builtin_amdgcn_global_load_async_to_lds_b128(
      (__attribute__((address_space(1))) v4i_*)g,
      (__attribute__((address_space(3))) v4i_*)l, 0, 0);
#else
  asm volatile("global_load_async_to_lds_b128 %0, %1, off"
               :: "v"((unsigned int)(unsigned long long)l), "v"(g) : "memory");
#endif
}
__device__ __forceinline__ void wait_async0() {
#if __has_builtin(__builtin_amdgcn_s_wait_asynccnt)
  __builtin_amdgcn_s_wait_asynccnt(0);
#else
  asm volatile("s_wait_asynccnt 0x0" ::: "memory");
#endif
}

// ---------------- conversion kernels ----------------

// each thread: 8 floats in -> 4 packed u32 out
__global__ void cvt_x(const float* __restrict__ x, unsigned int* __restrict__ out, int n4) {
  int i = blockIdx.x * blockDim.x + threadIdx.x;
  if (i >= n4) return;
  const float4* x4 = (const float4*)x;
  float4 a = x4[2 * i], b = x4[2 * i + 1];
  uint4 r;
  r.x = pack2(a.x, a.y); r.y = pack2(a.z, a.w);
  r.z = pack2(b.x, b.y); r.w = pack2(b.z, b.w);
  ((uint4*)out)[i] = r;
}

// w: [512][64] f32 -> wt: [64][512] bf16 (column of w becomes contiguous row)
__global__ void cvt_wt(const float* __restrict__ w, unsigned short* __restrict__ wt) {
  int tid = blockIdx.x * blockDim.x + threadIdx.x;   // 64*512 threads
  int n = tid >> 9, k = tid & 511;
  wt[tid] = f2bf(w[k * DKV + n]);
}

// ---------------- QKV projection GEMM ----------------
// One wave -> 16x64 output tile. A = 16x32 bf16 rows of x, B = 32x16 bf16 from w^T.
__global__ void __launch_bounds__(256) qkv_gemm(
    const unsigned int* __restrict__ xb,    // bf16 pairs, [B*S][256]
    const unsigned int* __restrict__ wt,    // bf16 pairs, [64][256]
    const float* __restrict__ bias,
    unsigned short* __restrict__ outb,      // bf16
    int transposed, float scale) {
  int wave  = (blockIdx.x * blockDim.x + threadIdx.x) >> 5;
  int lane  = threadIdx.x & 31;
  int laneL = lane & 15, hi = lane >> 4;
  int b       = wave >> 8;          // 256 tiles per batch
  int rowbase = (wave & 255) << 4;

  const unsigned int* xrow = xb + (size_t)((b * SQ) + rowbase + laneL) * (DM / 2);
  const unsigned int* wcol = wt + (size_t)laneL * (DM / 2);

  v8f acc[4] = {};
  for (int k0 = 0; k0 < DM; k0 += 32) {
    // gather ALL fragments for this k-step first, then issue 4 WMMAs back-to-back
    v8u av;
    int abase = (k0 >> 1) + hi * 8;
#pragma unroll
    for (int j = 0; j < 8; ++j)
      av[j] = xrow[abase + j];

    v8u bw[4];
#pragma unroll
    for (int n = 0; n < 4; ++n) {
      const unsigned int* wc = wcol + (size_t)(n * 16) * (DM / 2) + (k0 >> 1) + hi * 8;
#pragma unroll
      for (int j = 0; j < 8; ++j) bw[n][j] = wc[j];
    }

    v16bf A = __builtin_bit_cast(v16bf, av);
#pragma unroll
    for (int n = 0; n < 4; ++n)
      acc[n] = __builtin_amdgcn_wmma_f32_16x16x32_bf16(
          false, A, false, __builtin_bit_cast(v16bf, bw[n]), (short)0, acc[n], false, false);
  }

#pragma unroll
  for (int n = 0; n < 4; ++n) {
    int col = n * 16 + laneL;
    float bb = bias[col];
#pragma unroll
    for (int r = 0; r < 8; ++r) {
      int row = rowbase + r + hi * 8;     // C layout: lanes16-31 hold rows r+8
      float val = (acc[n][r] + bb) * scale;
      if (transposed)
        outb[(size_t)(b * DKV + col) * SQ + row] = f2bf(val);   // v^T: [b][dv][s]
      else
        outb[((size_t)(b * SQ) + row) * DKV + col] = f2bf(val);
    }
  }
}

// ---------------- flash attention ----------------
// Block = 8 waves, one batch. K/V tiles async-staged into LDS once per block,
// double-buffered so the ASYNC copy overlaps the WMMA work on the previous tile.
#define KSTRIDE 36   // 32 data dwords + pad (16B-aligned, 16 distinct banks)
#define VSTRIDE 20   // 16 data dwords + pad
#define PSTRIDE 20   // 16 data dwords + pad

__global__ void __launch_bounds__(256) attn(
    const unsigned int* __restrict__ qb,    // bf16 pairs [B*S][32], pre-scaled by 1/sqrt(dk)
    const unsigned int* __restrict__ kb,    // bf16 pairs [B*S][32]
    const unsigned int* __restrict__ vtb,   // bf16 pairs [B*64][2048]  (v transposed)
    float* __restrict__ out) {
  __shared__ unsigned int Kbuf[2][32 * KSTRIDE];
  __shared__ unsigned int Vbuf[2][64 * VSTRIDE];
  __shared__ unsigned int Pb[8][16 * PSTRIDE];

  int t     = threadIdx.x;
  int w     = t >> 5;
  int lane  = t & 31;
  int laneL = lane & 15, hi = lane >> 4;
  int wave  = blockIdx.x * 8 + w;
  int b     = wave >> 8;
  int qbase = (wave & 255) << 4;

  const unsigned int* kgb = kb + (size_t)(b * SQ) * (DKV / 2);
  const unsigned int* vgb = vtb + (size_t)(b * DKV) * (SQ / 2);

  // cooperative async staging: 256 threads x (16B K + 16B V) per tile
  int krow = t >> 3, kchunk = (t & 7) * 4;   // K: 32 rows x 32 dwords
  int vrow = t >> 2, vchunk = (t & 3) * 4;   // V: 64 rows x 16 dwords

  // prime buffer 0 with tile s0 = 0
  async16(kgb + (size_t)krow * (DKV / 2) + kchunk, &Kbuf[0][krow * KSTRIDE + kchunk]);
  async16(vgb + (size_t)vrow * (SQ / 2) + vchunk, &Vbuf[0][vrow * VSTRIDE + vchunk]);

  // Q fragments (features 0-31 and 32-63), A layout
  const unsigned int* qrow = qb + (size_t)(b * SQ + qbase + laneL) * (DKV / 2);
  v8u aq0, aq1;
#pragma unroll
  for (int j = 0; j < 8; ++j) {
    int off = j + hi * 8;
    aq0[j] = qrow[off];
    aq1[j] = qrow[16 + off];
  }
  v16bf Aq0 = __builtin_bit_cast(v16bf, aq0);
  v16bf Aq1 = __builtin_bit_cast(v16bf, aq1);

  float m[8], l[8];
#pragma unroll
  for (int r = 0; r < 8; ++r) { m[r] = -3.0e38f; l[r] = 0.f; }
  v8f o[4] = {};

  unsigned short* Ps = (unsigned short*)Pb[w];
  const float L2E = 1.4426950408889634f;

  for (int it = 0; it < SQ / 32; ++it) {
    int cur = it & 1;
    wait_async0();        // our async copies into buf[cur] are done
    __syncthreads();      // everyone's copies visible to all waves

    if (it + 1 < SQ / 32) {   // kick off next tile into the other buffer
      int s0n = (it + 1) * 32;
      async16(kgb + (size_t)(s0n + krow) * (DKV / 2) + kchunk,
              &Kbuf[cur ^ 1][krow * KSTRIDE + kchunk]);
      async16(vgb + (size_t)vrow * (SQ / 2) + (s0n >> 1) + vchunk,
              &Vbuf[cur ^ 1][vrow * VSTRIDE + vchunk]);
    }

    const unsigned int* Kt = Kbuf[cur];
    const unsigned int* Vt = Vbuf[cur];

    // ---- scores S = Q K^T for 32 keys: gather 4 LDS fragments, then 4 WMMAs ----
    v8u sk[4];
#pragma unroll
    for (int nt = 0; nt < 2; ++nt)
#pragma unroll
      for (int f = 0; f < 2; ++f)
#pragma unroll
        for (int j = 0; j < 8; ++j)
          sk[nt * 2 + f][j] = Kt[(nt * 16 + laneL) * KSTRIDE + f * 16 + hi * 8 + j];

    v8f c[2] = {};
    c[0] = __builtin_amdgcn_wmma_f32_16x16x32_bf16(false, Aq0, false,
            __builtin_bit_cast(v16bf, sk[0]), (short)0, c[0], false, false);
    c[0] = __builtin_amdgcn_wmma_f32_16x16x32_bf16(false, Aq1, false,
            __builtin_bit_cast(v16bf, sk[1]), (short)0, c[0], false, false);
    c[1] = __builtin_amdgcn_wmma_f32_16x16x32_bf16(false, Aq0, false,
            __builtin_bit_cast(v16bf, sk[2]), (short)0, c[1], false, false);
    c[1] = __builtin_amdgcn_wmma_f32_16x16x32_bf16(false, Aq1, false,
            __builtin_bit_cast(v16bf, sk[3]), (short)0, c[1], false, false);

    // ---- online softmax over the 32 new columns ----
#pragma unroll
    for (int r = 0; r < 8; ++r) {
      float s0v = c[0][r], s1v = c[1][r];
      float mx = fmaxf(s0v, s1v);
#pragma unroll
      for (int d = 1; d < 16; d <<= 1) mx = fmaxf(mx, __shfl_xor(mx, d, 32));
      float mn = fmaxf(m[r], mx);
      float p0 = exp2f((s0v - mn) * L2E);
      float p1 = exp2f((s1v - mn) * L2E);
      float al = exp2f((m[r] - mn) * L2E);
      float rs = p0 + p1;
#pragma unroll
      for (int d = 1; d < 16; d <<= 1) rs += __shfl_xor(rs, d, 32);
      l[r] = l[r] * al + rs;
      m[r] = mn;
#pragma unroll
      for (int f = 0; f < 4; ++f) o[f][r] *= al;
      int row = r + hi * 8;
      Ps[row * (2 * PSTRIDE) + laneL]      = f2bf(p0);
      Ps[row * (2 * PSTRIDE) + 16 + laneL] = f2bf(p1);
    }
    __syncthreads();

    // ---- reload P in A layout + V fragments, then 4 WMMAs ----
    v8u ap;
    const unsigned int* Pu = Pb[w];
#pragma unroll
    for (int j = 0; j < 8; ++j)
      ap[j] = Pu[laneL * PSTRIDE + hi * 8 + j];
    v16bf Ap = __builtin_bit_cast(v16bf, ap);

    v8u sv[4];
#pragma unroll
    for (int f = 0; f < 4; ++f)
#pragma unroll
      for (int j = 0; j < 8; ++j)
        sv[f][j] = Vt[(f * 16 + laneL) * VSTRIDE + hi * 8 + j];

#pragma unroll
    for (int f = 0; f < 4; ++f)
      o[f] = __builtin_amdgcn_wmma_f32_16x16x32_bf16(false, Ap, false,
              __builtin_bit_cast(v16bf, sv[f]), (short)0, o[f], false, false);

    __syncthreads();   // protect P buffer + K/V double-buffer reuse
  }

  // ---- normalize and write out f32 ----
#pragma unroll
  for (int f = 0; f < 4; ++f) {
    int col = f * 16 + laneL;
#pragma unroll
    for (int r = 0; r < 8; ++r) {
      int row = qbase + r + hi * 8;
      out[((size_t)(b * SQ) + row) * DKV + col] = o[f][r] / l[r];
    }
  }
}

// ---------------- launch ----------------

extern "C" void kernel_launch(void* const* d_in, const int* in_sizes, int n_in,
                              void* d_out, int out_size, void* d_ws, size_t ws_size,
                              hipStream_t stream) {
  const float* x  = (const float*)d_in[0];
  const float* wq = (const float*)d_in[1];
  const float* bq = (const float*)d_in[2];
  const float* wk = (const float*)d_in[3];
  const float* bk = (const float*)d_in[4];
  const float* wv = (const float*)d_in[5];
  const float* bv = (const float*)d_in[6];

  char* ws = (char*)d_ws;
  unsigned int*   xb  = (unsigned int*)(ws);                 // 16 MB  bf16 x
  unsigned short* qb  = (unsigned short*)(ws + 16777216);    // 2 MB   bf16 q (pre-scaled)
  unsigned short* kb  = (unsigned short*)(ws + 18874368);    // 2 MB   bf16 k
  unsigned short* vtb = (unsigned short*)(ws + 20971520);    // 2 MB   bf16 v^T
  unsigned short* wqt = (unsigned short*)(ws + 23068672);    // 64 KB  w_q^T
  unsigned short* wkt = (unsigned short*)(ws + 23134208);    // 64 KB  w_k^T
  unsigned short* wvt = (unsigned short*)(ws + 23199744);    // 64 KB  w_v^T

  cvt_x<<<4096, 256, 0, stream>>>(x, xb, BQ * SQ * DM / 8);
  cvt_wt<<<128, 256, 0, stream>>>(wq, wqt);
  cvt_wt<<<128, 256, 0, stream>>>(wk, wkt);
  cvt_wt<<<128, 256, 0, stream>>>(wv, wvt);

  // 1024 tiles of 16 rows x 64 cols; 8 waves/block -> 128 blocks
  qkv_gemm<<<128, 256, 0, stream>>>(xb, (const unsigned int*)wqt, bq, qb,  0, 0.125f);
  qkv_gemm<<<128, 256, 0, stream>>>(xb, (const unsigned int*)wkt, bk, kb,  0, 1.0f);
  qkv_gemm<<<128, 256, 0, stream>>>(xb, (const unsigned int*)wvt, bv, vtb, 1, 1.0f);

  attn<<<128, 256, 0, stream>>>((const unsigned int*)qb, (const unsigned int*)kb,
                                (const unsigned int*)vtb, (float*)d_out);
}
